// MetaDiscreteTimeTrajTypeGRUNet_36266703847745
// MI455X (gfx1250) — matrix operand order repaired and checked by reading, not compiled, not run on hardware
//
#include <hip/hip_runtime.h>
#include <hip/hip_bf16.h>

#define B_ 32
#define T_ 48
#define H_ 256
#define G3 768            // 3*H
#define E_ 128
#define TD_ 48
#define DIN_PAD 192       // 176 padded to multiple of 32
#define OUTN 40000
#define NOUT 40048
#define MROWS (B_ * T_)   // 1536

typedef __attribute__((ext_vector_type(16))) __bf16 v16bf;
typedef __attribute__((ext_vector_type(8)))  float  v8f;

union BfPack { v16bf v; unsigned int d[8]; };

__device__ __forceinline__ unsigned short f2bf(float f) {
  unsigned int x = __float_as_uint(f);
  unsigned int r = x + 0x7FFFu + ((x >> 16) & 1u);   // round-to-nearest-even
  return (unsigned short)(r >> 16);
}

// A-matrix 16x32 bf16 tile (ISA 7.12.2): lanes 0-15 hold K {0..7,16..23},
// lanes 16-31 hold K {8..15,24..31}; 2 halfs per VGPR -> dword loads.
__device__ __forceinline__ v16bf load_a_tile(const unsigned short* base, int ldk,
                                             int m0, int k0, int lane) {
  BfPack a;
  const unsigned short* row = base + (size_t)(m0 + (lane & 15)) * ldk;
  int kb = (lane >> 4) * 8;
#pragma unroll
  for (int i = 0; i < 4; ++i)
    a.d[i] = *(const unsigned int*)(row + k0 + kb + 2 * i);
#pragma unroll
  for (int i = 0; i < 4; ++i)
    a.d[4 + i] = *(const unsigned int*)(row + k0 + 16 + kb + 2 * i);
  return a.v;
}

// B-matrix 32x16 bf16 tile: lane = column n; lanes 0-15 hold K {0..7,16..23},
// lanes 16-31 hold K {8..15,24..31}. W is row-major [N][K] so column n of B is row n of W.
__device__ __forceinline__ v16bf load_b_tile(const unsigned short* W, int ldk,
                                             int n0, int k0, int lane) {
  BfPack b;
  const unsigned short* row = W + (size_t)(n0 + (lane & 15)) * ldk;
  int kb = (lane >> 4) * 8;
#pragma unroll
  for (int i = 0; i < 8; ++i)
    b.d[i] = *(const unsigned int*)(row + k0 + kb + 2 * i + (i >= 4 ? 8 : 0));
  return b.v;
}

// ---------------- conversion / setup kernels ----------------

__global__ void cvt_f32_bf16(const float* __restrict__ s, unsigned short* __restrict__ d, int n) {
  int i = blockIdx.x * blockDim.x + threadIdx.x;
  if (i < n) d[i] = f2bf(s[i]);
}

__global__ void cvt_pad_wih0(const float* __restrict__ s, unsigned short* __restrict__ d) {
  int i = blockIdx.x * blockDim.x + threadIdx.x;   // 768*192
  if (i >= G3 * DIN_PAD) return;
  int r = i / DIN_PAD, k = i % DIN_PAD;
  d[i] = (k < 176) ? f2bf(s[r * 176 + k]) : (unsigned short)0;
}

__global__ void build_x(const int* __restrict__ locs, const int* __restrict__ times,
                        const float* __restrict__ embed, unsigned short* __restrict__ xbf) {
  int i = blockIdx.x * blockDim.x + threadIdx.x;   // rows (t*B+b) x DIN_PAD
  if (i >= MROWS * DIN_PAD) return;
  int rr = i / DIN_PAD, k = i % DIN_PAD;
  int t = rr / B_, b = rr % B_;
  float v = 0.f;
  if (k < E_)            v = embed[(size_t)locs[b * T_ + t] * E_ + k];
  else if (k < E_ + TD_) v = (times[b * T_ + t] == (k - E_)) ? 1.f : 0.f;
  xbf[i] = f2bf(v);
}

// h0 = traj_table[labels].reshape(L, -1, H) -- raw reshape, flat-index preserving.
__global__ void init_h0(const float* __restrict__ traj, const int* __restrict__ labels,
                        float* __restrict__ h0) {
  int f = blockIdx.x * blockDim.x + threadIdx.x;
  if (f >= 2 * B_ * H_) return;
  int b = f >> 9;          // f / 512
  int c = f & 511;         // f % 512
  h0[f] = traj[(size_t)labels[b] * 512 + c];
}

// ---------------- generic WMMA bf16 GEMM: out = A[M,K] * W[N,K]^T + bias ----------------
// block = 256 threads (8 waves): 32(M) x 256(N) tile; each wave: 16 x 64.
// No control flow in the K loop: boundary tiles are clamped to column N-16
// (redundant in-bounds compute, EXEC stays all-1s for WMMA) and discarded at store.
// mode 0: out row-major [M,N] fp32.  mode 1: FC store with (t*B+b)->(b*T+t) row
// permutation, cols <40000 to loc region, >=40000 to tm region.
__global__ void gemm_bf16_wmma(const unsigned short* __restrict__ A,
                               const unsigned short* __restrict__ W,
                               const float* __restrict__ bias,
                               float* __restrict__ outp,
                               float* __restrict__ out_tm,
                               int N, int K, int mode) {
  int lane = threadIdx.x & 31;
  int wave = threadIdx.x >> 5;
  int m0 = blockIdx.y * 32 + (wave & 1) * 16;
  int nbase = blockIdx.x * 256 + (wave >> 1) * 64;

  // clamped (always in-bounds) tile origins; N is a multiple of 16
  int n0c[4];
#pragma unroll
  for (int j = 0; j < 4; ++j) {
    int n0 = nbase + j * 16;
    n0c[j] = (n0 <= N - 16) ? n0 : (N - 16);
  }

  v8f acc[4];
#pragma unroll
  for (int j = 0; j < 4; ++j) {
    float bv = bias[n0c[j] + (lane & 15)];
#pragma unroll
    for (int r = 0; r < 8; ++r) acc[j][r] = bv;
  }

  for (int k0 = 0; k0 < K; k0 += 32) {
    v16bf a = load_a_tile(A, K, m0, k0, lane);
#pragma unroll
    for (int j = 0; j < 4; ++j) {
      v16bf b = load_b_tile(W, K, n0c[j], k0, lane);
      acc[j] = __builtin_amdgcn_wmma_f32_16x16x32_bf16(
          false, a, false, b, (short)0, acc[j], false, false);
    }
  }

#pragma unroll
  for (int j = 0; j < 4; ++j) {
    if (nbase + j * 16 >= N) continue;   // wave-uniform discard, outside hot loop
    int n = n0c[j] + (lane & 15);
#pragma unroll
    for (int r = 0; r < 8; ++r) {
      int m = m0 + r + 8 * (lane >> 4);
      float v = acc[j][r];
      if (mode == 0) {
        outp[(size_t)m * N + n] = v;
      } else {
        int t = m >> 5, b = m & 31;      // A rows are t*B+b (B=32)
        int rout = b * T_ + t;
        if (n < OUTN) outp[(size_t)rout * OUTN + n] = v;
        else          out_tm[(size_t)rout * TD_ + (n - OUTN)] = v;
      }
    }
  }
}

// ---------------- persistent single-WGP GRU scan ----------------
// 512 threads = 16 waves. LDS: gbuf [32][768] f32 (96KB) + hbf [32][256] bf16 (16KB).
// fp32 hidden state lives in registers (16 elems/thread).
__global__ void gru_scan(const float* __restrict__ xp,            // [T*B, 768]
                         const unsigned short* __restrict__ Whh,  // [768,256] bf16
                         const float* __restrict__ bhh,           // [768]
                         const float* __restrict__ h0,            // [32,256] f32
                         unsigned short* __restrict__ yout,       // [T*B,256] bf16
                         int relu_out) {
  extern __shared__ char smem[];
  float* gbuf = (float*)smem;                                   // 32*768 f32
  unsigned short* hbf = (unsigned short*)(smem + B_ * G3 * 4);  // 32*256 bf16
  int tid = threadIdx.x;
  int lane = tid & 31;
  int wave = tid >> 5;

  float hreg[16];
#pragma unroll
  for (int i = 0; i < 16; ++i) {
    int e = tid + i * 512;
    float v = h0[e];
    hreg[i] = v;
    hbf[e] = f2bf(v);
  }
  __syncthreads();

  for (int t = 0; t < T_; ++t) {
    // phase 1: gbuf = h @ Whh^T + bhh  (96 tiles of 16x16, 6 per wave)
#pragma unroll
    for (int i = 0; i < 6; ++i) {
      int tt = wave + 16 * i;
      int m0 = (tt & 1) * 16;
      int n0 = (tt >> 1) * 16;
      v8f acc;
      float bv = bhh[n0 + (lane & 15)];
#pragma unroll
      for (int r = 0; r < 8; ++r) acc[r] = bv;
#pragma unroll
      for (int k0 = 0; k0 < H_; k0 += 32) {
        v16bf a = load_a_tile(hbf, H_, m0, k0, lane);   // ds_load from LDS
        v16bf b = load_b_tile(Whh, H_, n0, k0, lane);   // global (L2-hot)
        acc = __builtin_amdgcn_wmma_f32_16x16x32_bf16(
            false, a, false, b, (short)0, acc, false, false);
      }
      int n = n0 + (lane & 15);
#pragma unroll
      for (int r = 0; r < 8; ++r) {
        int m = m0 + r + 8 * (lane >> 4);
        gbuf[m * G3 + n] = acc[r];
      }
    }
    __syncthreads();

    // phase 2: gates (PyTorch GRU: r,z,n ordering)
    const float* xpt = xp + (size_t)t * B_ * G3;
#pragma unroll
    for (int i = 0; i < 16; ++i) {
      int e = tid + i * 512;
      int b = e >> 8;
      int j = e & 255;
      const float* xr = xpt + b * G3;
      const float* gr = gbuf + b * G3;
      float r = 1.f / (1.f + __expf(-(xr[j] + gr[j])));
      float z = 1.f / (1.f + __expf(-(xr[H_ + j] + gr[H_ + j])));
      float n = tanhf(xr[2 * H_ + j] + r * gr[2 * H_ + j]);
      float hn = (1.f - z) * n + z * hreg[i];
      hreg[i] = hn;
      hbf[e] = f2bf(hn);
      float yv = relu_out ? fmaxf(hn, 0.f) : hn;
      yout[((size_t)t * B_ + b) * H_ + j] = f2bf(yv);
    }
    __syncthreads();
  }
}

// ---------------- log-softmax ----------------

__global__ void lsm_reduce(const float* __restrict__ loc, const float* __restrict__ tm,
                           float* __restrict__ c_loc, float* __restrict__ c_tm) {
  __shared__ float red[256];
  int r = blockIdx.x;
  int tid = threadIdx.x;
  const float* row = loc + (size_t)r * OUTN;
  float m = -3.0e38f;
  for (int n = tid; n < OUTN; n += 256) m = fmaxf(m, row[n]);
  red[tid] = m; __syncthreads();
  for (int s = 128; s > 0; s >>= 1) { if (tid < s) red[tid] = fmaxf(red[tid], red[tid + s]); __syncthreads(); }
  float M = red[0]; __syncthreads();
  float ss = 0.f;
  for (int n = tid; n < OUTN; n += 256) ss += __expf(row[n] - M);
  red[tid] = ss; __syncthreads();
  for (int s = 128; s > 0; s >>= 1) { if (tid < s) red[tid] += red[tid + s]; __syncthreads(); }
  if (tid == 0) c_loc[r] = M + __logf(red[0]);
  __syncthreads();

  const float* trow = tm + (size_t)r * TD_;
  float mv = (tid < TD_) ? trow[tid] : -3.0e38f;
  red[tid] = mv; __syncthreads();
  for (int s = 128; s > 0; s >>= 1) { if (tid < s) red[tid] = fmaxf(red[tid], red[tid + s]); __syncthreads(); }
  float Mt = red[0]; __syncthreads();
  float st = (tid < TD_) ? __expf(trow[tid] - Mt) : 0.f;
  red[tid] = st; __syncthreads();
  for (int s = 128; s > 0; s >>= 1) { if (tid < s) red[tid] += red[tid + s]; __syncthreads(); }
  if (tid == 0) c_tm[r] = Mt + __logf(red[0]);
}

__global__ void lsm_apply_loc(float* __restrict__ loc, const float* __restrict__ c_loc) {
  int r = blockIdx.y;
  int n = blockIdx.x * 256 + threadIdx.x;
  if (n < OUTN) loc[(size_t)r * OUTN + n] -= c_loc[r];
}

__global__ void lsm_apply_tm(float* __restrict__ tm, const float* __restrict__ c_tm) {
  int i = blockIdx.x * 256 + threadIdx.x;
  if (i < MROWS * TD_) tm[i] -= c_tm[i / TD_];
}

// ---------------- host launch ----------------

extern "C" void kernel_launch(void* const* d_in, const int* in_sizes, int n_in,
                              void* d_out, int out_size, void* d_ws, size_t ws_size,
                              hipStream_t stream) {
  const int*   locations = (const int*)d_in[0];
  const int*   times     = (const int*)d_in[1];
  const int*   labels    = (const int*)d_in[2];
  const float* embed     = (const float*)d_in[3];
  const float* traj      = (const float*)d_in[4];
  const float* fc_w      = (const float*)d_in[5];
  const float* fc_b      = (const float*)d_in[6];
  const float* w_ih0     = (const float*)d_in[7];
  const float* w_hh0     = (const float*)d_in[8];
  const float* b_ih0     = (const float*)d_in[9];
  const float* b_hh0     = (const float*)d_in[10];
  const float* w_ih1     = (const float*)d_in[11];
  const float* w_hh1     = (const float*)d_in[12];
  const float* b_ih1     = (const float*)d_in[13];
  const float* b_hh1     = (const float*)d_in[14];

  char* ws = (char*)d_ws;
  size_t off = 0;
  auto alloc = [&](size_t bytes) -> void* {
    void* p = ws + off;
    off = (off + bytes + 255) & ~(size_t)255;
    return p;
  };

  unsigned short* x_bf  = (unsigned short*)alloc((size_t)MROWS * DIN_PAD * 2);
  unsigned short* wih0b = (unsigned short*)alloc((size_t)G3 * DIN_PAD * 2);
  unsigned short* whh0b = (unsigned short*)alloc((size_t)G3 * H_ * 2);
  unsigned short* wih1b = (unsigned short*)alloc((size_t)G3 * H_ * 2);
  unsigned short* whh1b = (unsigned short*)alloc((size_t)G3 * H_ * 2);
  unsigned short* fcwb  = (unsigned short*)alloc((size_t)NOUT * H_ * 2);
  float* xp0 = (float*)alloc((size_t)MROWS * G3 * 4);
  float* xp1 = (float*)alloc((size_t)MROWS * G3 * 4);
  unsigned short* y0b = (unsigned short*)alloc((size_t)MROWS * H_ * 2);
  unsigned short* y1b = (unsigned short*)alloc((size_t)MROWS * H_ * 2);
  float* h0w  = (float*)alloc((size_t)2 * B_ * H_ * 4);
  float* c_loc = (float*)alloc((size_t)MROWS * 4);
  float* c_tm  = (float*)alloc((size_t)MROWS * 4);

  // weight conversions (fp32 -> bf16; L2-resident afterwards)
  cvt_pad_wih0<<<(G3 * DIN_PAD + 255) / 256, 256, 0, stream>>>(w_ih0, wih0b);
  cvt_f32_bf16<<<(G3 * H_ + 255) / 256, 256, 0, stream>>>(w_hh0, whh0b, G3 * H_);
  cvt_f32_bf16<<<(G3 * H_ + 255) / 256, 256, 0, stream>>>(w_ih1, wih1b, G3 * H_);
  cvt_f32_bf16<<<(G3 * H_ + 255) / 256, 256, 0, stream>>>(w_hh1, whh1b, G3 * H_);
  cvt_f32_bf16<<<(NOUT * H_ + 255) / 256, 256, 0, stream>>>(fc_w, fcwb, NOUT * H_);

  build_x<<<(MROWS * DIN_PAD + 255) / 256, 256, 0, stream>>>(locations, times, embed, x_bf);
  init_h0<<<(2 * B_ * H_ + 255) / 256, 256, 0, stream>>>(traj, labels, h0w);

  // xp0 = x @ w_ih0^T + b_ih0   [1536 x 768], K=192
  gemm_bf16_wmma<<<dim3(3, 48), 256, 0, stream>>>(x_bf, wih0b, b_ih0, xp0, nullptr,
                                                  G3, DIN_PAD, 0);
  // layer-0 recurrence (single WGP, LDS-resident state)
  size_t smem = (size_t)B_ * G3 * 4 + (size_t)B_ * H_ * 2;   // 96KB + 16KB
  gru_scan<<<1, 512, smem, stream>>>(xp0, whh0b, b_hh0, h0w, y0b, 0);
  // xp1 = y0 @ w_ih1^T + b_ih1
  gemm_bf16_wmma<<<dim3(3, 48), 256, 0, stream>>>(y0b, wih1b, b_ih1, xp1, nullptr,
                                                  G3, H_, 0);
  // layer-1 recurrence, ReLU fused into bf16 output
  gru_scan<<<1, 512, smem, stream>>>(xp1, whh1b, b_hh1, h0w + B_ * H_, y1b, 1);

  // FC: logits straight into d_out (loc + tm regions), row-permuted to [B,T,...]
  float* out     = (float*)d_out;
  float* out_loc = out;
  float* out_tm  = out + (size_t)MROWS * OUTN;
  gemm_bf16_wmma<<<dim3((NOUT + 255) / 256, 48), 256, 0, stream>>>(
      y1b, fcwb, fc_b, out_loc, out_tm, NOUT, H_, 1);

  // log-softmax in place
  lsm_reduce<<<MROWS, 256, 0, stream>>>(out_loc, out_tm, c_loc, c_tm);
  lsm_apply_loc<<<dim3((OUTN + 255) / 256, MROWS), 256, 0, stream>>>(out_loc, c_loc);
  lsm_apply_tm<<<(MROWS * TD_ + 255) / 256, 256, 0, stream>>>(out_tm, c_tm);
}
